// MambaBlock_89103391523247
// MI455X (gfx1250) — matrix operand order repaired and checked by reading, not compiled, not run on hardware
//
#include <hip/hip_runtime.h>
#include <hip/hip_bf16.h>

// ---------------------------------------------------------------------------
// MambaBlock for MI455X (gfx1250, wave32, WMMA, async global->LDS copies)
//   U = x@B            : bf16 WMMA GEMM           (8192x1024x256)
//   scan (chunked)     : fp32 VALU, A in 256KB LDS (parallel chunks + carry)
//   Y = S@C^T + D      : bf16 WMMA GEMM           (8192x256x1024)
//   H = relu(Y@W1+b1)  : bf16 WMMA GEMM           (8192x1024x4096)
//   out = H@W2 + b2    : bf16 WMMA GEMM -> fp32   (8192x4096x1024)
// All B-side operands are prepacked to n-major [N][K] bf16 so both LDS tiles
// stage as contiguous 16B chunks via GLOBAL_LOAD_ASYNC_TO_LDS_B128.
// ---------------------------------------------------------------------------

typedef __attribute__((ext_vector_type(16))) __bf16 v16bf;
typedef __attribute__((ext_vector_type(8)))  __bf16 v8bf;
typedef __attribute__((ext_vector_type(8)))  float  v8f;
typedef int v4i __attribute__((vector_size(16)));

#define DMODEL 1024
#define DSTATE 256
#define DFF    4096
#define NBATCH 4
#define SEQLEN 2048
#define CHUNK  32
#define NCHUNK (SEQLEN / CHUNK)   // 64

// ---- gfx1250 async global->LDS copy (ASYNCcnt path) -----------------------
#if defined(__has_builtin)
#  if __has_builtin(__builtin_amdgcn_global_load_async_to_lds_b128)
#    define HAVE_ASYNC_COPY 1
#  endif
#  if __has_builtin(__builtin_amdgcn_s_wait_asynccnt)
#    define HAVE_WAIT_ASYNC 1
#  endif
#endif

using gv4i_t = __attribute__((address_space(1))) v4i;
using lv4i_t = __attribute__((address_space(3))) v4i;

__device__ __forceinline__ void copy16_g2l(const __bf16* g, __bf16* l) {
#if defined(HAVE_ASYNC_COPY)
#pragma clang diagnostic push
#pragma clang diagnostic ignored "-Wold-style-cast"
    __builtin_amdgcn_global_load_async_to_lds_b128((gv4i_t*)g, (lv4i_t*)l, 0, 0);
#pragma clang diagnostic pop
#else
    *(v8bf*)l = *(const v8bf*)g;
#endif
}

__device__ __forceinline__ void wait_copies() {
#if defined(HAVE_ASYNC_COPY)
#  if defined(HAVE_WAIT_ASYNC)
    __builtin_amdgcn_s_wait_asynccnt(0);
#  else
    asm volatile("s_wait_asynccnt 0x0" ::: "memory");
#  endif
#endif
}

// ---------------------------------------------------------------------------
// Conversions / prepack
// ---------------------------------------------------------------------------
__global__ void cvt_f32_bf16(const float* __restrict__ in, __bf16* __restrict__ out, int n) {
    int i = blockIdx.x * blockDim.x + threadIdx.x;
    if (i < n) out[i] = (__bf16)in[i];
}

// in: [R][C] f32 row-major  ->  out: [C][R] bf16 (B-side prepack)
__global__ void transpose_cvt_bf16(const float* __restrict__ in, __bf16* __restrict__ out,
                                   int R, int C) {
    int o = blockIdx.x * blockDim.x + threadIdx.x;
    if (o >= R * C) return;
    int oc = o / R;          // original column
    int orr = o % R;         // original row
    out[o] = (__bf16)in[orr * C + oc];
}

// in: [R][C] bf16 -> out: [C][R] bf16 (matpow intermediate prepack)
__global__ void transpose_bf16(const __bf16* __restrict__ in, __bf16* __restrict__ out,
                               int R, int C) {
    int o = blockIdx.x * blockDim.x + threadIdx.x;
    if (o >= R * C) return;
    int oc = o / R;
    int orr = o % R;
    out[o] = in[orr * C + oc];
}

// ---------------------------------------------------------------------------
// bf16 WMMA GEMM:  out[M,N] = op(A[M,K] @ B + bias),  B prepacked [N][K] bf16.
// Block: 256 threads (8 waves), tile 128x128; wave tile 32x64 = 2x4 WMMA tiles.
// Double-buffered LDS, async global->LDS staging overlapped with WMMA.
// Requires M%128==0, N%128==0, K%32==0.
// ---------------------------------------------------------------------------
template<int RELU, int BIAS, int OUT_BF16>
__global__ __launch_bounds__(256)
void gemm_bf16(const __bf16* __restrict__ A, const __bf16* __restrict__ Bp,
               const float* __restrict__ bias, void* __restrict__ out,
               int M, int N, int K) {
    __shared__ __bf16 lA[2][128][32];   // [m][k] : 8 KB per buffer
    __shared__ __bf16 lB[2][128][32];   // [n][k] : 8 KB per buffer

    const int tid   = threadIdx.x;
    const int lane  = tid & 31;
    const int wid   = tid >> 5;
    const int lm    = lane & 15;     // 0..15
    const int half  = lane >> 4;     // 0..1
    const int waveM = wid & 3;       // 0..3 -> 32-row strip
    const int waveN = wid >> 2;      // 0..1 -> 64-col strip
    const int m0    = blockIdx.y * 128;
    const int n0    = blockIdx.x * 128;

    v8f acc[2][4];
    #pragma unroll
    for (int mi = 0; mi < 2; ++mi)
        #pragma unroll
        for (int ni = 0; ni < 4; ++ni)
            #pragma unroll
            for (int r = 0; r < 8; ++r) acc[mi][ni][r] = 0.0f;

    // stage one 128x32 A tile + 128x32 B tile (1024 x 16B chunks, 4/thread)
    auto stage = [&](int buf, int kk) {
        #pragma unroll
        for (int i = 0; i < 2; ++i) {
            int c   = tid + i * 256;          // 0..511, no divergence
            int row = c >> 2;
            int off = (c & 3) << 3;
            copy16_g2l(&A[(size_t)(m0 + row) * K + kk + off],  &lA[buf][row][off]);
            copy16_g2l(&Bp[(size_t)(n0 + row) * K + kk + off], &lB[buf][row][off]);
        }
    };

    const int nk = K / 32;
    stage(0, 0);
    for (int ki = 0; ki < nk; ++ki) {
        const int buf = ki & 1;
        wait_copies();       // own async copies into `buf` complete
        __syncthreads();     // everyone's copies complete; prev compute done

        if (ki + 1 < nk) stage(buf ^ 1, (ki + 1) * 32);
        if (ki + 2 < nk) {   // warm L2 two tiles ahead (global_prefetch_b8)
            __builtin_prefetch(&A[(size_t)(m0 + (tid >> 1)) * K + (ki + 2) * 32 + ((tid & 1) << 4)], 0, 1);
            __builtin_prefetch(&Bp[(size_t)(n0 + (tid >> 1)) * K + (ki + 2) * 32 + ((tid & 1) << 4)], 0, 1);
        }

        // ---- fragments per gfx1250 VGPR layouts ----
        v16bf afrag[2], bfrag[4];
        #pragma unroll
        for (int mi = 0; mi < 2; ++mi) {
            int row = waveM * 32 + mi * 16 + lm;
            // A 16x32: elems 0..7 -> K = half*8+i ; elems 8..15 -> K = 16+half*8+i
            v8bf lo = *(const v8bf*)&lA[buf][row][half * 8];
            v8bf hi = *(const v8bf*)&lA[buf][row][16 + half * 8];
            afrag[mi] = __builtin_shufflevector(lo, hi,
                0, 1, 2, 3, 4, 5, 6, 7, 8, 9, 10, 11, 12, 13, 14, 15);
        }
        #pragma unroll
        for (int ni = 0; ni < 4; ++ni) {
            int col = waveN * 64 + ni * 16 + lm;
            // B 32x16: elems 0..7 -> K = half*8+i ; elems 8..15 -> K = 16+half*8+i
            v8bf blo = *(const v8bf*)&lB[buf][col][half * 8];
            v8bf bhi = *(const v8bf*)&lB[buf][col][16 + half * 8];
            bfrag[ni] = __builtin_shufflevector(blo, bhi,
                0, 1, 2, 3, 4, 5, 6, 7, 8, 9, 10, 11, 12, 13, 14, 15);
        }
        #pragma unroll
        for (int mi = 0; mi < 2; ++mi)
            #pragma unroll
            for (int ni = 0; ni < 4; ++ni)
                acc[mi][ni] = __builtin_amdgcn_wmma_f32_16x16x32_bf16(
                    false, afrag[mi], false, bfrag[ni],
                    (short)0, acc[mi][ni], false, false);
    }

    // ---- epilogue: C/D layout -> row = r + 8*half, col = lm ----
    #pragma unroll
    for (int mi = 0; mi < 2; ++mi)
        #pragma unroll
        for (int ni = 0; ni < 4; ++ni)
            #pragma unroll
            for (int r = 0; r < 8; ++r) {
                int row = m0 + waveM * 32 + mi * 16 + half * 8 + r;
                int col = n0 + waveN * 64 + ni * 16 + lm;
                float v = acc[mi][ni][r];
                if (BIAS) v += bias[col];
                if (RELU) v = v > 0.0f ? v : 0.0f;
                if (OUT_BF16) ((__bf16*)out)[(size_t)row * N + col] = (__bf16)v;
                else          ((float*)out)[(size_t)row * N + col] = v;
            }
}

// ---------------------------------------------------------------------------
// Chunked scan. One block per (chunk g, batch b); A (256KB fp32) resident in
// LDS (WGP has 320KB). Thread n owns state column n; conflict-free banks.
// PHASE 1: zero-init, emit chunk-end state E.  PHASE 3: carry-init, emit all
// per-step states as bf16 (feeds the S@C^T WMMA GEMM).
// ---------------------------------------------------------------------------
template<int PHASE>
__global__ __launch_bounds__(256)
void scan_chunk(const float* __restrict__ A, const float* __restrict__ U,
                const float* __restrict__ carry, float* __restrict__ E,
                __bf16* __restrict__ St) {
    extern __shared__ float sm[];
    float* sA = sm;                       // 65536 floats = 256 KB
    float* st = sm + DSTATE * DSTATE;     // 2 x 256 double-buffered state

    const int n = threadIdx.x;            // 0..255
    const int g = blockIdx.x;             // chunk
    const int b = blockIdx.y;             // batch

    for (int i = n; i < DSTATE * DSTATE; i += 256) sA[i] = A[i];
    st[n] = (PHASE == 3) ? carry[(g * NBATCH + b) * DSTATE + n] : 0.0f;

    int cur = 0;
    const size_t urow0 = (size_t)(b * SEQLEN + g * CHUNK) * DSTATE;
    for (int t = 0; t < CHUNK; ++t) {
        __syncthreads();
        float accv = U[urow0 + (size_t)t * DSTATE + n];
        const float* sp = st + cur * DSTATE;
        #pragma unroll 8
        for (int k = 0; k < DSTATE; ++k)
            accv = fmaf(sp[k], sA[k * DSTATE + n], accv);
        st[(cur ^ 1) * DSTATE + n] = accv;
        cur ^= 1;
        if (PHASE == 3) St[urow0 + (size_t)t * DSTATE + n] = (__bf16)accv;
    }
    if (PHASE == 1) E[(g * NBATCH + b) * DSTATE + n] = st[cur * DSTATE + n];
}

// ---------------------------------------------------------------------------
// Sequential carry combine over 64 chunks: c_g = c_{g-1} @ A^CHUNK + E_{g-1}
// ---------------------------------------------------------------------------
__global__ __launch_bounds__(256)
void carry_seq(const __bf16* __restrict__ AL, const float* __restrict__ E,
               float* __restrict__ carry) {
    extern __shared__ float sm[];
    float* sA = sm;                       // 65536
    float* cp = sm + DSTATE * DSTATE;     // NBATCH * 256

    const int n = threadIdx.x;
    for (int i = n; i < DSTATE * DSTATE; i += 256) sA[i] = (float)AL[i];
    for (int b = 0; b < NBATCH; ++b) {
        cp[b * DSTATE + n] = 0.0f;
        carry[(0 * NBATCH + b) * DSTATE + n] = 0.0f;
    }
    __syncthreads();
    for (int g = 1; g < NCHUNK; ++g) {
        float tmp[NBATCH];
        for (int b = 0; b < NBATCH; ++b) {
            float accv = E[((g - 1) * NBATCH + b) * DSTATE + n];
            for (int k = 0; k < DSTATE; ++k)
                accv = fmaf(cp[b * DSTATE + k], sA[k * DSTATE + n], accv);
            tmp[b] = accv;
        }
        __syncthreads();
        for (int b = 0; b < NBATCH; ++b) {
            cp[b * DSTATE + n] = tmp[b];
            carry[(g * NBATCH + b) * DSTATE + n] = tmp[b];
        }
        __syncthreads();
    }
}

// ---------------------------------------------------------------------------
extern "C" void kernel_launch(void* const* d_in, const int* in_sizes, int n_in,
                              void* d_out, int out_size, void* d_ws, size_t ws_size,
                              hipStream_t stream) {
    (void)in_sizes; (void)n_in; (void)out_size; (void)ws_size;

    const float* x  = (const float*)d_in[0];
    const float* A  = (const float*)d_in[1];
    const float* Bw = (const float*)d_in[2];
    const float* Cw = (const float*)d_in[3];
    const float* Dw = (const float*)d_in[4];
    const float* W1 = (const float*)d_in[5];
    const float* b1 = (const float*)d_in[6];
    const float* W2 = (const float*)d_in[7];
    const float* b2 = (const float*)d_in[8];
    float* out = (float*)d_out;

    const int MROWS = NBATCH * SEQLEN;    // 8192

    // ---- workspace carve (256B aligned) ----
    char* ws = (char*)d_ws;
    size_t off = 0;
    auto take = [&](size_t bytes) -> char* {
        char* p = ws + off;
        off += (bytes + 255) & ~(size_t)255;
        return p;
    };
    __bf16* xbf   = (__bf16*)take((size_t)MROWS * DMODEL * 2);
    __bf16* BwP   = (__bf16*)take((size_t)DSTATE * DMODEL * 2);   // [256][1024]
    __bf16* W1P   = (__bf16*)take((size_t)DFF * DMODEL * 2);      // [4096][1024]
    __bf16* W2P   = (__bf16*)take((size_t)DMODEL * DFF * 2);      // [1024][4096]
    __bf16* CbfP  = (__bf16*)take((size_t)DMODEL * DSTATE * 2);   // C as-is = prepacked C^T
    __bf16* Arow  = (__bf16*)take((size_t)DSTATE * DSTATE * 2);
    __bf16* Acol  = (__bf16*)take((size_t)DSTATE * DSTATE * 2);
    __bf16* powR0 = (__bf16*)take((size_t)DSTATE * DSTATE * 2);
    __bf16* powR1 = (__bf16*)take((size_t)DSTATE * DSTATE * 2);
    __bf16* powC0 = (__bf16*)take((size_t)DSTATE * DSTATE * 2);
    __bf16* powC1 = (__bf16*)take((size_t)DSTATE * DSTATE * 2);
    float*  U     = (float*) take((size_t)MROWS * DSTATE * 4);
    float*  Ebuf  = (float*) take((size_t)NCHUNK * NBATCH * DSTATE * 4);
    float*  Cry   = (float*) take((size_t)NCHUNK * NBATCH * DSTATE * 4);
    __bf16* St    = (__bf16*)take((size_t)MROWS * DSTATE * 2);
    __bf16* Ybf   = (__bf16*)take((size_t)MROWS * DMODEL * 2);
    __bf16* Hbf   = (__bf16*)take((size_t)MROWS * DFF * 2);

    // ---- one-time conversions / B-side prepacks ----
    {
        int n;
        n = MROWS * DMODEL;  cvt_f32_bf16<<<(n + 255) / 256, 256, 0, stream>>>(x, xbf, n);
        n = DSTATE * DSTATE; cvt_f32_bf16<<<(n + 255) / 256, 256, 0, stream>>>(A, Arow, n);
        n = DMODEL * DSTATE; cvt_f32_bf16<<<(n + 255) / 256, 256, 0, stream>>>(Cw, CbfP, n);
        n = DMODEL * DSTATE; transpose_cvt_bf16<<<(n + 255) / 256, 256, 0, stream>>>(Bw, BwP, DMODEL, DSTATE);
        n = DSTATE * DSTATE; transpose_cvt_bf16<<<(n + 255) / 256, 256, 0, stream>>>(A, Acol, DSTATE, DSTATE);
        n = DMODEL * DFF;    transpose_cvt_bf16<<<(n + 255) / 256, 256, 0, stream>>>(W1, W1P, DMODEL, DFF);
        n = DFF * DMODEL;    transpose_cvt_bf16<<<(n + 255) / 256, 256, 0, stream>>>(W2, W2P, DFF, DMODEL);
    }

    // ---- U = x @ B (fp32 out, feeds scan) ----
    gemm_bf16<0, 0, 0><<<dim3(DSTATE / 128, MROWS / 128), 256, 0, stream>>>(
        xbf, BwP, nullptr, U, MROWS, DSTATE, DMODEL);

    // ---- A^CHUNK via 5 WMMA squarings (A^2..A^32), keep row+prepack forms ----
    {
        __bf16* curR = Arow;
        __bf16* curC = Acol;
        __bf16* rowsb[2] = {powR0, powR1};
        __bf16* colsb[2] = {powC0, powC1};
        const int n = DSTATE * DSTATE;
        for (int i = 0; i < 5; ++i) {
            __bf16* nr = rowsb[i & 1];
            __bf16* nc = colsb[i & 1];
            gemm_bf16<0, 0, 1><<<dim3(DSTATE / 128, DSTATE / 128), 256, 0, stream>>>(
                curR, curC, nullptr, nr, DSTATE, DSTATE, DSTATE);
            transpose_bf16<<<(n + 255) / 256, 256, 0, stream>>>(nr, nc, DSTATE, DSTATE);
            curR = nr; curC = nc;
        }
        // curR == A^32 row-major bf16
        // ---- chunked scan ----
        const size_t scan_sm  = (size_t)(DSTATE * DSTATE + 2 * DSTATE) * 4;
        const size_t carry_sm = (size_t)(DSTATE * DSTATE + NBATCH * DSTATE) * 4;
        scan_chunk<1><<<dim3(NCHUNK, NBATCH), 256, scan_sm, stream>>>(A, U, nullptr, Ebuf, nullptr);
        carry_seq<<<1, 256, carry_sm, stream>>>(curR, Ebuf, Cry);
        scan_chunk<3><<<dim3(NCHUNK, NBATCH), 256, scan_sm, stream>>>(A, U, Cry, nullptr, St);
    }

    // ---- Y = St @ C^T + D (bf16 out) ----
    gemm_bf16<0, 1, 1><<<dim3(DMODEL / 128, MROWS / 128), 256, 0, stream>>>(
        St, CbfP, Dw, Ybf, MROWS, DMODEL, DSTATE);

    // ---- H = relu(Y @ W1 + b1) (bf16 out) ----
    gemm_bf16<1, 1, 1><<<dim3(DFF / 128, MROWS / 128), 256, 0, stream>>>(
        Ybf, W1P, b1, Hbf, MROWS, DFF, DMODEL);

    // ---- out = H @ W2 + b2 (fp32 out) ----
    gemm_bf16<0, 1, 0><<<dim3(DMODEL / 128, MROWS / 128), 256, 0, stream>>>(
        Hbf, W2P, b2, out, MROWS, DMODEL, DFF);
}
